// SearchTransformer_35751307771927
// MI455X (gfx1250) — matrix operand order repaired and checked by reading, not compiled, not run on hardware
//
#include <hip/hip_runtime.h>
#include <hip/hip_bf16.h>

typedef __attribute__((ext_vector_type(16))) __bf16 v16bf;
typedef __attribute__((ext_vector_type(8)))  __bf16 v8bf;
typedef __attribute__((ext_vector_type(8)))  float  v8f;

#define NB   4            // batch
#define CIN  256          // channels of lv3 (4*C)
#define HP   48           // lv3 spatial
#define LPOS (HP * HP)    // 2304 columns
#define DDIM (CIN * 9)    // 2304 unfold depth

__device__ __forceinline__ unsigned short f32_to_bf16(float f) {
  unsigned int u = __float_as_uint(f);
  u += 0x7FFFu + ((u >> 16) & 1u);         // round-to-nearest-even
  return (unsigned short)(u >> 16);
}

// ---------------------------------------------------------------------------
// Kernel 1: 3x3/pad1/stride1 unfold + L2-normalize + bf16 pack.
// One 256-thread block per (tensor z, batch y, column x); thread = channel.
// Output layout: dst[((n*2304 + pos) * 2304) + c*9 + j]  (contiguous d per column)
// ---------------------------------------------------------------------------
__global__ __launch_bounds__(256)
void unfold_norm_bf16(const float* __restrict__ lr, const float* __restrict__ refp,
                      unsigned short* __restrict__ Qb, unsigned short* __restrict__ Kb) {
  const int pos = blockIdx.x;            // 0..2303
  const int n   = blockIdx.y;            // 0..3
  const int y = pos / HP, x = pos % HP;
  const int c = threadIdx.x;             // 0..255
  const float* src = (blockIdx.z == 0) ? lr : refp;
  unsigned short* dst = (blockIdx.z == 0) ? Qb : Kb;
  const float* plane = src + (size_t)(n * CIN + c) * (HP * HP);

  float v[9];
  float ss = 0.f;
#pragma unroll
  for (int dy = 0; dy < 3; ++dy)
#pragma unroll
    for (int dx = 0; dx < 3; ++dx) {
      int rr = y - 1 + dy, cc = x - 1 + dx;
      float t = (rr >= 0 && rr < HP && cc >= 0 && cc < HP) ? plane[rr * HP + cc] : 0.f;
      v[dy * 3 + dx] = t;
      ss += t * t;
    }

  __shared__ float red[256];
  red[c] = ss;
  __syncthreads();
  for (int off = 128; off > 0; off >>= 1) {
    if (c < off) red[c] += red[c + off];
    __syncthreads();
  }
  const float scale = 1.f / fmaxf(sqrtf(red[0]), 1e-12f);

  unsigned short* outp = dst + ((size_t)n * LPOS + pos) * DDIM + c * 9;
#pragma unroll
  for (int j = 0; j < 9; ++j) outp[j] = f32_to_bf16(v[j] * scale);
}

// ---------------------------------------------------------------------------
// Kernel 2: rel = K^T * Q (per batch, 2304x2304x2304) fused with argmax over
// the key dimension. One wave handles 16 query columns; loops over 144 L-tiles
// of 16 key rows, each accumulated via 72x v_wmma_f32_16x16x32_bf16.
//
// Fragment layouts (CDNA5 wave32, 16-bit):
//   A (16x32, rows=l, row-major in d): lane<16 -> row l0+lane, d+0..15;
//     lane>=16 -> row l0+lane-16, d+16..31 (same k order as B).
//   B (32x16, cols=m, column-major in d): lane holds 16 contiguous d of its
//     column; lanes 16..31 take the d+16..31 half.
//   C (f32 16x16): VGPR r, lane<16 -> (M=r, N=lane); lane>=16 -> (M=r+8).
// ---------------------------------------------------------------------------
__global__ __launch_bounds__(128)
void relmax_wmma(const unsigned short* __restrict__ Qb,
                 const unsigned short* __restrict__ Kb,
                 float* __restrict__ sOut, int* __restrict__ idxOut) {
  const int lane = threadIdx.x & 31;
  const int wave = threadIdx.x >> 5;
  const int g = blockIdx.x * 4 + wave;            // 576 waves total, no tail
  const int n = g / (LPOS / 16);
  const int mTile = g % (LPOS / 16);
  const int m0 = mTile * 16;
  const int half = lane >> 4;                     // 0/1 lane half
  const int lr = lane & 15;

  const __bf16* Q = (const __bf16*)Qb + (size_t)n * LPOS * DDIM;
  const __bf16* K = (const __bf16*)Kb + (size_t)n * LPOS * DDIM;

  const __bf16* bcol = Q + (size_t)(m0 + lr) * DDIM + half * 16;

  float bestVal = -3.402823466e38f;
  int   bestIdx = 0;

  for (int l0 = 0; l0 < LPOS; l0 += 16) {
    const __bf16* arow = K + (size_t)(l0 + lr) * DDIM + half * 16;
    v8f acc = {};
#pragma unroll 4
    for (int d0 = 0; d0 < DDIM; d0 += 32) {
      v8bf alo = *(const v8bf*)(arow + d0);        // K 0..7  (or 16..23)
      v8bf ahi = *(const v8bf*)(arow + d0 + 8);    // K 8..15 (or 24..31)
      v16bf a;
#pragma unroll
      for (int i = 0; i < 8; ++i) { a[i] = alo[i]; a[i + 8] = ahi[i]; }
      v16bf b = *(const v16bf*)(bcol + d0);        // 16 contiguous d
      acc = __builtin_amdgcn_wmma_f32_16x16x32_bf16(
          false, a, false, b, (short)0, acc, false, false);
    }
    // Branch-free argmax update (keeps EXEC all-ones around WMMA).
#pragma unroll
    for (int r = 0; r < 8; ++r) {
      float vv = acc[r];
      int li = l0 + half * 8 + r;                  // ascending l per lane
      bool upd = vv > bestVal;                     // strict > keeps first max
      bestVal = upd ? vv : bestVal;
      bestIdx = upd ? li : bestIdx;
    }
  }

  // Merge the two lane halves that share the same query column.
  float ov = __shfl_xor(bestVal, 16, 32);
  int   oi = __shfl_xor(bestIdx, 16, 32);
  bool take = (ov > bestVal) || (ov == bestVal && oi < bestIdx);
  bestVal = take ? ov : bestVal;
  bestIdx = take ? oi : bestIdx;
  if (half == 0) {
    sOut[n * LPOS + m0 + lr]   = bestVal;          // s written directly
    idxOut[n * LPOS + m0 + lr] = bestIdx;
  }
}

// ---------------------------------------------------------------------------
// Kernel 3: gather winning key windows + fold (transposed conv accumulation).
// out[n,c,h,w] = (1/9) * sum over <=9 covering windows at query pos (yq,xq)
//               of ref[n, c, py*stride-pad+dy, px*stride-pad+dx], p=idx[yq,xq].
// ---------------------------------------------------------------------------
__global__ __launch_bounds__(256)
void gather_fold(const float* __restrict__ ref, const int* __restrict__ idx,
                 float* __restrict__ out, int C, int Hin, int k, int pad,
                 int stride, int Hout) {
  const long long gid = (long long)blockIdx.x * blockDim.x + threadIdx.x;
  const long long total = (long long)NB * C * Hout * Hout;
  if (gid >= total) return;
  const int w = (int)(gid % Hout);
  const int h = (int)((gid / Hout) % Hout);
  const int c = (int)((gid / ((long long)Hout * Hout)) % C);
  const int n = (int)(gid / ((long long)C * Hout * Hout));
  const int* idxn = idx + n * LPOS;
  const float* plane = ref + (size_t)(n * C + c) * Hin * Hin;

  float acc = 0.f;
  for (int dy = 0; dy < k; ++dy) {
    int ty = h + pad - dy;
    if (ty < 0 || (ty % stride) != 0) continue;
    int yq = ty / stride;
    if (yq >= HP) continue;
    for (int dx = 0; dx < k; ++dx) {
      int tx = w + pad - dx;
      if (tx < 0 || (tx % stride) != 0) continue;
      int xq = tx / stride;
      if (xq >= HP) continue;
      int p = idxn[yq * HP + xq];
      int py = p / HP, px = p % HP;
      int rr = py * stride - pad + dy;
      int cc = px * stride - pad + dx;
      if (rr >= 0 && rr < Hin && cc >= 0 && cc < Hin)
        acc += plane[rr * Hin + cc];
    }
  }
  out[gid] = acc / 9.0f;
}

// ---------------------------------------------------------------------------
extern "C" void kernel_launch(void* const* d_in, const int* in_sizes, int n_in,
                              void* d_out, int out_size, void* d_ws, size_t ws_size,
                              hipStream_t stream) {
  (void)in_sizes; (void)n_in; (void)out_size; (void)ws_size;
  const float* lr   = (const float*)d_in[0];   // (4,256,48,48)
  const float* refp = (const float*)d_in[1];   // (4,256,48,48)
  const float* ref1 = (const float*)d_in[2];   // (4,64,192,192)
  const float* ref2 = (const float*)d_in[3];   // (4,128,96,96)
  const float* ref3 = (const float*)d_in[4];   // (4,256,48,48)

  // Workspace: Q bf16 (42.5MB) | K bf16 (42.5MB) | idx i32 (36.9KB)
  const size_t qkElems = (size_t)NB * LPOS * DDIM;
  unsigned short* Qb = (unsigned short*)d_ws;
  unsigned short* Kb = Qb + qkElems;
  int* idx = (int*)(Kb + qkElems);

  // Output: s | t_lv3 | t_lv2 | t_lv1 (flat, f32)
  float* s  = (float*)d_out;
  float* t3 = s  + (size_t)NB * LPOS;
  float* t2 = t3 + (size_t)NB * 256 * 48 * 48;
  float* t1 = t2 + (size_t)NB * 128 * 96 * 96;

  unfold_norm_bf16<<<dim3(LPOS, NB, 2), 256, 0, stream>>>(lr, refp, Qb, Kb);
  relmax_wmma<<<dim3((NB * (LPOS / 16)) / 4), 128, 0, stream>>>(Qb, Kb, s, idx);

  gather_fold<<<dim3((NB * 256 * 48 * 48  + 255) / 256), 256, 0, stream>>>(
      ref3, idx, t3, 256,  48,  3, 1, 1,  48);
  gather_fold<<<dim3((NB * 128 * 96 * 96  + 255) / 256), 256, 0, stream>>>(
      ref2, idx, t2, 128,  96,  6, 2, 2,  96);
  gather_fold<<<dim3((NB * 64 * 192 * 192 + 255) / 256), 256, 0, stream>>>(
      ref1, idx, t1,  64, 192, 12, 4, 4, 192);
}
